// ResNet_DeformAttention_9637906612256
// MI455X (gfx1250) — compile-verified, matrix operand order checked by reading, not run on hardware
//
#include <hip/hip_runtime.h>

// ---------------- CDNA5 WMMA types ----------------
typedef __attribute__((ext_vector_type(16))) __bf16 v16bf;
typedef __attribute__((ext_vector_type(8)))  __bf16 v8bf;
typedef __attribute__((ext_vector_type(2)))  __bf16 v2bf;
typedef __attribute__((ext_vector_type(8)))  float  v8f;

__device__ __forceinline__ __bf16 to_bf16(float f) { return (__bf16)f; }

// Load a 16-element bf16 fragment: two contiguous 8-element runs at p and p+16.
// Matches the ISA 16-bit A/B operand layout (K runs {kb..kb+7, kb+16..kb+23}).
__device__ __forceinline__ v16bf ldfrag(const __bf16* p) {
  const v8bf lo = *(const v8bf*)(p);
  const v8bf hi = *(const v8bf*)(p + 16);
  v16bf r;
#pragma unroll
  for (int i = 0; i < 8; ++i) { r[i] = lo[i]; r[i + 8] = hi[i]; }
  return r;
}

__device__ __forceinline__ v8f wmma_bf16(v16bf a, v16bf b, v8f c) {
  // (neg_a, A, neg_b, B, c_mod, C, reuse_a, reuse_b)
  return __builtin_amdgcn_wmma_f32_16x16x32_bf16(false, a, false, b, (short)0, c,
                                                 false, false);
}

// Async global->LDS copy of one 16-byte chunk (per lane), tracked by ASYNCcnt.
__device__ __forceinline__ void async_copy_b128(void* lds_dst, const void* gsrc) {
  const unsigned loff = (unsigned)(unsigned long long)lds_dst;  // LDS byte offset
  const unsigned long long ga = (unsigned long long)gsrc;
  asm volatile("global_load_async_to_lds_b128 %0, %1, off"
               :: "v"(loff), "v"(ga) : "memory");
}
__device__ __forceinline__ void async_wait0() {
  asm volatile("s_wait_asynccnt 0x0" ::: "memory");
}

// ---------------- Generic GEMM: C = A(MxK) * B(KxN) + bias, batched over B/C --
// block = 256 threads (8 waves). Tile 64(M) x 128(N) x 32(K).
__global__ void gemm_bf16_wmma(const float* __restrict__ A,
                               const float* __restrict__ Bg,
                               const float* __restrict__ bias,
                               float* __restrict__ Cg,
                               int M, int N, int K, long strideB, long strideC) {
  __shared__ alignas(16) __bf16 As[64 * 32];   // [row][k]
  __shared__ alignas(16) __bf16 Bs[128 * 32];  // [n][k] (transposed)

  const int tid  = threadIdx.x;
  const int wave = tid >> 5;
  const int lane = tid & 31;
  const int hf   = lane >> 4;   // which 16-lane half
  const int l15  = lane & 15;
  const int m0 = blockIdx.y * 64;
  const int n0 = blockIdx.x * 128;
  const int rg = (wave & 3) * 16;     // row group inside tile
  const int cb = (wave >> 2) * 64;    // col base inside tile (4 subtiles)

  const float* Bb = Bg + (long)blockIdx.z * strideB;
  float*       Cb = Cg + (long)blockIdx.z * strideC;

  v8f acc[4] = {{}, {}, {}, {}};

  for (int k0 = 0; k0 < K; k0 += 32) {
    __syncthreads();
    {   // A tile: 64x32; 8 consecutive k per thread -> one ds_store_b128
      const int row = tid >> 2, kk = (tid & 3) * 8;
      const float* src = A + (long)(m0 + row) * K + k0 + kk;
      __builtin_prefetch(src + 32, 0, 1);
      v8bf a8;
#pragma unroll
      for (int j = 0; j < 8; ++j) a8[j] = to_bf16(src[j]);
      *(v8bf*)(&As[row * 32 + kk]) = a8;
    }
    {   // B tile: 32x128 -> Bs[n][k]; k-pair x 8n per thread -> packed b32 stores
      const int kp = tid >> 4;            // 0..15 -> k = 2*kp
      const int nn = (tid & 15) * 8;
      const float* s0p = Bb + (long)(k0 + 2 * kp) * N + n0 + nn;
      const float* s1p = s0p + N;
#pragma unroll
      for (int j = 0; j < 8; ++j) {
        v2bf pk; pk[0] = to_bf16(s0p[j]); pk[1] = to_bf16(s1p[j]);
        *(v2bf*)(&Bs[(nn + j) * 32 + 2 * kp]) = pk;
      }
    }
    __syncthreads();

    const v16bf af = ldfrag(&As[(rg + l15) * 32 + hf * 8]);
#pragma unroll
    for (int i = 0; i < 4; ++i) {
      const v16bf bfr = ldfrag(&Bs[(cb + i * 16 + l15) * 32 + hf * 8]);
      acc[i] = wmma_bf16(af, bfr, acc[i]);
    }
  }

  float bvals[8];
#pragma unroll
  for (int r = 0; r < 8; ++r) bvals[r] = bias[m0 + rg + hf * 8 + r];

#pragma unroll
  for (int i = 0; i < 4; ++i) {
    const int col = n0 + cb + i * 16 + l15;
#pragma unroll
    for (int r = 0; r < 8; ++r) {
      const int row = m0 + rg + hf * 8 + r;
      Cb[(long)row * N + col] = acc[i][r] + bvals[r];
    }
  }
}

// ---------------- Depthwise 3x3x3 stride-2 conv on q -------------------------
// q viewed as (B*4, 64, 8, 32, 32) -> off (B*4, 64, 4, 16, 16)
__global__ void conv_dw_kernel(const float* __restrict__ q,
                               const float* __restrict__ Wdw,
                               const float* __restrict__ bdw,
                               float* __restrict__ off) {
  const int i = blockIdx.x * 256 + threadIdx.x;      // 524288 total
  const int wp = i & 15, hp = (i >> 4) & 15, tp = (i >> 8) & 3;
  const int c = (i >> 10) & 63, bg = i >> 16;
  const int b = bg >> 2, g = bg & 3;
  const float* base = q + (long)(b * 256 + g * 64 + c) * 8192;
  float acc = bdw[c];
  for (int kz = 0; kz < 3; ++kz) {
    const int z = 2 * tp - 1 + kz; if (z < 0 || z >= 8) continue;
    for (int ky = 0; ky < 3; ++ky) {
      const int y = 2 * hp - 1 + ky; if (y < 0 || y >= 32) continue;
      for (int kx = 0; kx < 3; ++kx) {
        const int xc = 2 * wp - 1 + kx; if (xc < 0 || xc >= 32) continue;
        acc += Wdw[c * 27 + kz * 9 + ky * 3 + kx] * base[z * 1024 + y * 32 + xc];
      }
    }
  }
  off[(long)bg * 65536 + c * 1024 + tp * 256 + hp * 16 + wp] = acc;
}

// ---------------- LayerNorm + GELU + 1x1->3 + tanh + sampling coords ---------
__global__ void offset_kernel(const float* __restrict__ off,
                              const float* __restrict__ lnw,
                              const float* __restrict__ lnb,
                              const float* __restrict__ Wout,
                              float* __restrict__ coords) {
  const int i = blockIdx.x * 256 + threadIdx.x;      // 8192 total
  const int s = i & 1023, bg = i >> 10;
  const float* base = off + (long)bg * 65536 + s;    // channel stride 1024
  float sum = 0.f, sq = 0.f;
  for (int c = 0; c < 64; ++c) { const float v = base[c * 1024]; sum += v; sq += v * v; }
  const float mu  = sum * (1.f / 64.f);
  const float var = sq * (1.f / 64.f) - mu * mu;
  const float inv = rsqrtf(var + 1e-5f);
  float a0 = 0.f, a1 = 0.f, a2 = 0.f;
  for (int c = 0; c < 64; ++c) {
    const float v  = (base[c * 1024] - mu) * inv * lnw[c] + lnb[c];
    const float gl = 0.5f * v * (1.f + erff(v * 0.70710678118654752f));
    a0 += Wout[c] * gl; a1 += Wout[64 + c] * gl; a2 += Wout[128 + c] * gl;
  }
  const float oz = tanhf(a0) * 0.5f;      // min(1, 2/4)
  const float oy = tanhf(a1) * 0.125f;    // min(1, 2/16)
  const float ox = tanhf(a2) * 0.125f;
  const int wp = s & 15, hp = (s >> 4) & 15, tp = s >> 8;
  const float pz = oz + ((tp + 0.5f) / 4.f)  * 2.f - 1.f;
  const float py = oy + ((hp + 0.5f) / 16.f) * 2.f - 1.f;
  const float px = ox + ((wp + 0.5f) / 16.f) * 2.f - 1.f;
  coords[(long)i * 3 + 0] = (px + 1.f) * 0.5f * 31.f;   // ix
  coords[(long)i * 3 + 1] = (py + 1.f) * 0.5f * 31.f;   // iy
  coords[(long)i * 3 + 2] = (pz + 1.f) * 0.5f * 7.f;    // iz
}

// ---------------- Trilinear grid sample of x ---------------------------------
__global__ void sample_kernel(const float* __restrict__ x,
                              const float* __restrict__ coords,
                              float* __restrict__ xs) {
  const int i = blockIdx.x * 256 + threadIdx.x;      // 524288 total
  const int s = i & 1023, c = (i >> 10) & 63, bg = i >> 16;
  const int b = bg >> 2, g = bg & 3;
  const float* cp = coords + ((long)bg * 1024 + s) * 3;
  const float ix = cp[0], iy = cp[1], iz = cp[2];
  const float x0f = floorf(ix), y0f = floorf(iy), z0f = floorf(iz);
  const float fx = ix - x0f, fy = iy - y0f, fz = iz - z0f;
  const int x0 = (int)x0f, y0 = (int)y0f, z0 = (int)z0f;
  const float* base = x + (long)(b * 256 + g * 64 + c) * 8192;
  float acc = 0.f;
#pragma unroll
  for (int dz = 0; dz < 2; ++dz) {
    const int zz = z0 + dz; const float wz = dz ? fz : 1.f - fz;
#pragma unroll
    for (int dy = 0; dy < 2; ++dy) {
      const int yy = y0 + dy; const float wy = dy ? fy : 1.f - fy;
#pragma unroll
      for (int dx = 0; dx < 2; ++dx) {
        const int xx = x0 + dx; const float wx = dx ? fx : 1.f - fx;
        const bool valid = (xx >= 0) && (xx < 32) && (yy >= 0) && (yy < 32) &&
                           (zz >= 0) && (zz < 8);
        const int zc = min(max(zz, 0), 7), yc = min(max(yy, 0), 31),
                  xc = min(max(xx, 0), 31);
        acc += base[zc * 1024 + yc * 32 + xc] * (wx * wy * wz) * (valid ? 1.f : 0.f);
      }
    }
  }
  xs[(long)(b * 256 + g * 64 + c) * 1024 + s] = acc;
}

// ---------------- Repack k/v into bf16 LDS-ready layouts ---------------------
// Kp[bh][n][c] (n-major, for score B-frags); Vp[bh][c][n] (c-major, for PV B-frags)
__global__ void repack_kv(const float* __restrict__ kb, const float* __restrict__ vb,
                          __bf16* __restrict__ Kp, __bf16* __restrict__ Vp) {
  const int bh = blockIdx.x, b = bh >> 3, h = bh & 7;
  const long cbase = (long)b * 256 + h * 32;
  const int c = threadIdx.x >> 3, nb = (threadIdx.x & 7) * 128;
  const float* ks = kb + (cbase + c) * 1024 + nb;
  const float* vs = vb + (cbase + c) * 1024 + nb;
  __bf16* kp = Kp + (long)bh * 32768;
  __bf16* vp = Vp + (long)bh * 32768;
  for (int j = 0; j < 128; ++j) {
    kp[(nb + j) * 32 + c] = to_bf16(ks[j]);
    vp[c * 1024 + nb + j] = to_bf16(vs[j]);
  }
}

// ---------------- Fused flash attention (HC=32, S=1024) ----------------------
// grid(64, 16): 16 = (b,h); block 256 = 8 waves, each wave 16 query rows.
// K,V slice (32x1024 bf16, 64KB each) DMA'd into LDS via async global->LDS
// loads (ASYNCcnt); online softmax over 32-wide n-chunks; P transposed
// C-layout -> A-layout through per-wave LDS scratch (LDS in-order per wave).
__global__ void attn_kernel(const float* __restrict__ q,
                            const __bf16* __restrict__ Kp,
                            const __bf16* __restrict__ Vp,
                            float* __restrict__ o) {
  const int bh = blockIdx.y, b = bh >> 3, h = bh & 7;
  const int tid = threadIdx.x;
  const int wave = tid >> 5, lane = tid & 31, hf = lane >> 4, l15 = lane & 15;
  const int m0 = blockIdx.x * 128 + wave * 16;
  const long cbase = (long)b * 256 + h * 32;

  extern __shared__ char smem_raw[];
  __bf16* Ks = (__bf16*)smem_raw;   // [n][c]  : 1024*32
  __bf16* Vs = Ks + 1024 * 32;      // [c][n]  : 32*1024
  __bf16* Ps = Vs + 32 * 1024;      // [wave][16][32]

  {   // async DMA: 2 x 64KB global -> LDS, 16B per lane per issue
    const __bf16* kp = Kp + (long)bh * 32768;
    const __bf16* vp = Vp + (long)bh * 32768;
#pragma unroll
    for (int i = 0; i < 16; ++i) {
      const int t = tid + i * 256;                   // 0..4095 chunks of 8 bf16
      async_copy_b128(Ks + t * 8, kp + (long)t * 8);
    }
#pragma unroll
    for (int i = 0; i < 16; ++i) {
      const int t = tid + i * 256;
      async_copy_b128(Vs + t * 8, vp + (long)t * 8);
    }
    async_wait0();
  }
  __syncthreads();

  // Q fragment (A operand, 16 rows x 32 c), pre-scaled by 1/sqrt(HC)
  v16bf qf;
  {
    const float scale = 0.17677669529663687f;
    const long m = m0 + l15;
#pragma unroll
    for (int e = 0; e < 8; ++e) {
      const int c0 = hf * 8 + e, c1 = 16 + hf * 8 + e;
      qf[e]     = to_bf16(q[(cbase + c0) * 8192 + m] * scale);
      qf[e + 8] = to_bf16(q[(cbase + c1) * 8192 + m] * scale);
    }
  }

  v8f o0 = {}, o1 = {};
  float rmax[8], lsum[8];
#pragma unroll
  for (int r = 0; r < 8; ++r) { rmax[r] = -1e30f; lsum[r] = 0.f; }
  __bf16* Pw = Ps + wave * (16 * 32);

  for (int n0 = 0; n0 < 1024; n0 += 32) {
    const v16bf bk0 = ldfrag(&Ks[(n0 + l15) * 32 + hf * 8]);
    const v16bf bk1 = ldfrag(&Ks[(n0 + 16 + l15) * 32 + hf * 8]);
    v8f z = {};
    v8f s0 = wmma_bf16(qf, bk0, z);
    v8f s1 = wmma_bf16(qf, bk1, z);

#pragma unroll
    for (int r = 0; r < 8; ++r) {
      float cm = fmaxf(s0[r], s1[r]);
      cm = fmaxf(cm, __shfl_xor(cm, 1, 32));
      cm = fmaxf(cm, __shfl_xor(cm, 2, 32));
      cm = fmaxf(cm, __shfl_xor(cm, 4, 32));
      cm = fmaxf(cm, __shfl_xor(cm, 8, 32));
      const float nm    = fmaxf(rmax[r], cm);
      const float alpha = __expf(rmax[r] - nm);
      const float p0 = __expf(s0[r] - nm);
      const float p1 = __expf(s1[r] - nm);
      float rs = p0 + p1;
      rs += __shfl_xor(rs, 1, 32);
      rs += __shfl_xor(rs, 2, 32);
      rs += __shfl_xor(rs, 4, 32);
      rs += __shfl_xor(rs, 8, 32);
      lsum[r] = lsum[r] * alpha + rs;
      rmax[r] = nm;
      o0[r] *= alpha; o1[r] *= alpha;
      const int row = r + hf * 8;                 // C-layout row
      Pw[row * 32 + l15]      = to_bf16(p0);
      Pw[row * 32 + 16 + l15] = to_bf16(p1);
    }
    // LDS ops are in-order within a wave: safe to read back immediately.
    const v16bf pf  = ldfrag(&Pw[l15 * 32 + hf * 8]);
    const v16bf bv0 = ldfrag(&Vs[l15 * 1024 + n0 + hf * 8]);
    const v16bf bv1 = ldfrag(&Vs[(16 + l15) * 1024 + n0 + hf * 8]);
    o0 = wmma_bf16(pf, bv0, o0);
    o1 = wmma_bf16(pf, bv1, o1);
  }

#pragma unroll
  for (int r = 0; r < 8; ++r) {
    const float inv = 1.f / lsum[r];
    const long m = m0 + hf * 8 + r;
    o[(cbase + l15) * 8192 + m]      = o0[r] * inv;
    o[(cbase + 16 + l15) * 8192 + m] = o1[r] * inv;
  }
}

// ---------------- Host-side orchestration ------------------------------------
extern "C" void kernel_launch(void* const* d_in, const int* in_sizes, int n_in,
                              void* d_out, int out_size, void* d_ws, size_t ws_size,
                              hipStream_t stream) {
  const float* x    = (const float*)d_in[0];
  const float* Wq   = (const float*)d_in[1];
  const float* bq   = (const float*)d_in[2];
  const float* Wk   = (const float*)d_in[3];
  const float* bk   = (const float*)d_in[4];
  const float* Wv   = (const float*)d_in[5];
  const float* bv   = (const float*)d_in[6];
  const float* Wo   = (const float*)d_in[7];
  const float* bo   = (const float*)d_in[8];
  const float* Wdw  = (const float*)d_in[9];
  const float* bdw  = (const float*)d_in[10];
  const float* lnw  = (const float*)d_in[11];
  const float* lnb  = (const float*)d_in[12];
  const float* Wout = (const float*)d_in[13];
  float* out = (float*)d_out;

  float* ws     = (float*)d_ws;
  float* qb     = ws;                       // 2*256*8192
  float* attn_o = qb + 4194304;             // 2*256*8192
  float* offb   = attn_o + 4194304;         // 8*64*1024
  float* coords = offb + 524288;            // 8*1024*3
  float* xsb    = coords + 24576;           // 2*256*1024
  float* kb     = xsb + 524288;             // 2*256*1024
  float* vb     = kb + 524288;              // 2*256*1024
  __bf16* kp16  = (__bf16*)(vb + 524288);   // 16*1024*32 bf16 (1 MB)
  __bf16* vp16  = kp16 + 524288;            // 16*32*1024 bf16 (1 MB)

  const dim3 blk(256);

  // q = Wq * x + bq          (M=256, N=8192, K=256, batch 2)
  gemm_bf16_wmma<<<dim3(64, 4, 2), blk, 0, stream>>>(Wq, x, bq, qb,
                                                     256, 8192, 256, 2097152L, 2097152L);
  // offset branch
  conv_dw_kernel<<<2048, blk, 0, stream>>>(qb, Wdw, bdw, offb);
  offset_kernel<<<32, blk, 0, stream>>>(offb, lnw, lnb, Wout, coords);
  sample_kernel<<<2048, blk, 0, stream>>>(x, coords, xsb);
  // k/v projections          (M=256, N=1024, K=256, batch 2)
  gemm_bf16_wmma<<<dim3(8, 4, 2), blk, 0, stream>>>(Wk, xsb, bk, kb,
                                                    256, 1024, 256, 262144L, 262144L);
  gemm_bf16_wmma<<<dim3(8, 4, 2), blk, 0, stream>>>(Wv, xsb, bv, vb,
                                                    256, 1024, 256, 262144L, 262144L);
  // bf16 repack into LDS-ready layouts for the async DMA
  repack_kv<<<16, blk, 0, stream>>>(kb, vb, kp16, vp16);
  // fused attention: 136 KB dynamic LDS (fits CDNA5's 320 KB WGP LDS)
  const size_t smem = (size_t)(1024 * 32 + 32 * 1024 + 8 * 16 * 32) * sizeof(__bf16);
  attn_kernel<<<dim3(64, 16), blk, smem, stream>>>(qb, kp16, vp16, attn_o);
  // out = Wo * attn + bo
  gemm_bf16_wmma<<<dim3(64, 4, 2), blk, 0, stream>>>(Wo, attn_o, bo, out,
                                                     256, 8192, 256, 2097152L, 2097152L);
}
